// CrossModalAttentionBlock_26319559590532
// MI455X (gfx1250) — matrix operand order repaired and checked by reading, not compiled, not run on hardware
//
#include <hip/hip_runtime.h>
#include <hip/hip_bf16.h>

#define D_MODEL   1024
#define NUM_HEADS 16
#define HEAD_DIM  64
#define T_Q       1024
#define T_CTX     2048
#define BATCH     4
#define LN_EPS    1e-5f
#define Q_EPS     1e-5f
#define ATT_SCALE 0.125f   // 64^-0.5

typedef __attribute__((ext_vector_type(16))) __bf16 v16bf;
typedef __attribute__((ext_vector_type(8)))  float  v8f;
typedef __attribute__((ext_vector_type(4)))  int    v4i;

// CDNA5 async global->LDS path (ASYNCcnt), with compile-safe fallback.
#if defined(__has_builtin)
#  if __has_builtin(__builtin_amdgcn_global_load_async_to_lds_b128) && \
      __has_builtin(__builtin_amdgcn_s_wait_asynccnt)
#    define USE_ASYNC_LDS 1
#  endif
#endif
#ifndef USE_ASYNC_LDS
#  define USE_ASYNC_LDS 0
#endif

__device__ __forceinline__ v8f zero8() {
  v8f z = {0.f, 0.f, 0.f, 0.f, 0.f, 0.f, 0.f, 0.f};
  return z;
}

// ---- WMMA bf16 16x16x32 fragment loaders (wave32) ----
// A-matrix: lane-half h: VGPR0-3 = K kbase+h*8+{0..7}, VGPR4-7 = K kbase+16+h*8+{0..7}
__device__ __forceinline__ v16bf load_fragA(const __bf16* __restrict__ base, int stride, int kbase) {
  const int lane = (int)(threadIdx.x & 31);
  const int r = lane & 15, h = lane >> 4;
  const __bf16* p = base + (size_t)r * stride + kbase + h * 8;
  union { uint4 u[2]; v16bf v; } t;
  t.u[0] = *(const uint4*)(p);
  t.u[1] = *(const uint4*)(p + 16);
  return t.v;
}
// B-matrix: lane-half h: VGPR0-3 = K kbase+h*8+{0..7}, VGPR4-7 = K kbase+16+h*8+{0..7}; row n of [N,K] array.
__device__ __forceinline__ v16bf load_fragB(const __bf16* __restrict__ base, int stride, int kbase) {
  const int lane = (int)(threadIdx.x & 31);
  const int r = lane & 15, h = lane >> 4;
  const __bf16* p = base + (size_t)r * stride + kbase + h * 8;
  union { uint4 u[2]; v16bf v; } t;
  t.u[0] = *(const uint4*)(p);
  t.u[1] = *(const uint4*)(p + 16);
  return t.v;
}
__device__ __forceinline__ v8f wmma_bf16(v16bf a, v16bf b, v8f c) {
  return __builtin_amdgcn_wmma_f32_16x16x32_bf16(false, a, false, b, (short)0, c, false, false);
}

// ---------------- gamma = mean(|W|), deterministic 2-stage reduction ----------------
__global__ __launch_bounds__(256) void absmean_s1(const float* __restrict__ W0, const float* __restrict__ W1,
                                                  const float* __restrict__ W2, const float* __restrict__ W3,
                                                  float* __restrict__ part) {
  __shared__ float red[256];
  const int mi = blockIdx.y;
  const float* W = mi == 0 ? W0 : mi == 1 ? W1 : mi == 2 ? W2 : W3;
  const int N = D_MODEL * D_MODEL;
  float s = 0.f;
  for (int i = blockIdx.x * 256 + threadIdx.x; i < N; i += 256 * 256)
    s += fabsf(W[i]);
  red[threadIdx.x] = s;
  __syncthreads();
  for (int st = 128; st > 0; st >>= 1) {
    if ((int)threadIdx.x < st) red[threadIdx.x] += red[threadIdx.x + st];
    __syncthreads();
  }
  if (threadIdx.x == 0) part[mi * 256 + blockIdx.x] = red[0];
}

__global__ __launch_bounds__(256) void absmean_s2(const float* __restrict__ part, float* __restrict__ gamma) {
  __shared__ float red[256];
  red[threadIdx.x] = part[blockIdx.x * 256 + threadIdx.x];
  __syncthreads();
  for (int st = 128; st > 0; st >>= 1) {
    if ((int)threadIdx.x < st) red[threadIdx.x] += red[threadIdx.x + st];
    __syncthreads();
  }
  if (threadIdx.x == 0) gamma[blockIdx.x] = red[0] * (1.f / (float)(D_MODEL * D_MODEL));
}

// ---------------- ternary quantization: store {-1,0,+1} exactly in bf16 ----------------
__global__ __launch_bounds__(256) void quantize_kernel(const float* __restrict__ W0, const float* __restrict__ W1,
                                                       const float* __restrict__ W2, const float* __restrict__ W3,
                                                       const float* __restrict__ gamma,
                                                       __bf16* __restrict__ Q0, __bf16* __restrict__ Q1,
                                                       __bf16* __restrict__ Q2, __bf16* __restrict__ Q3) {
  const int mi = blockIdx.y;
  const float* W = mi == 0 ? W0 : mi == 1 ? W1 : mi == 2 ? W2 : W3;
  __bf16* Q = mi == 0 ? Q0 : mi == 1 ? Q1 : mi == 2 ? Q2 : Q3;
  const float inv = 1.f / (gamma[mi] + Q_EPS);
  const int N = D_MODEL * D_MODEL;
  for (int i = blockIdx.x * blockDim.x + threadIdx.x; i < N; i += gridDim.x * blockDim.x) {
    float t = rintf(W[i] * inv);
    t = fminf(1.f, fmaxf(-1.f, t));
    Q[i] = (__bf16)t;
  }
}

// ---------------- layernorm + cast to bf16 (one block per row) ----------------
__global__ __launch_bounds__(256) void ln_cast_kernel(const float* __restrict__ x, const float* __restrict__ g,
                                                      const float* __restrict__ be, __bf16* __restrict__ out) {
  __shared__ float red[256];
  const int row = blockIdx.x;
  const int tid = threadIdx.x;
  const float* xr = x + (size_t)row * D_MODEL;
  float v4[4];
  float s = 0.f;
#pragma unroll
  for (int i = 0; i < 4; ++i) { v4[i] = xr[tid + 256 * i]; s += v4[i]; }
  red[tid] = s;
  __syncthreads();
  for (int st = 128; st > 0; st >>= 1) { if (tid < st) red[tid] += red[tid + st]; __syncthreads(); }
  const float mu = red[0] * (1.f / (float)D_MODEL);
  __syncthreads();
  float s2 = 0.f;
#pragma unroll
  for (int i = 0; i < 4; ++i) { float d = v4[i] - mu; s2 += d * d; }
  red[tid] = s2;
  __syncthreads();
  for (int st = 128; st > 0; st >>= 1) { if (tid < st) red[tid] += red[tid + st]; __syncthreads(); }
  const float inv = rsqrtf(red[0] * (1.f / (float)D_MODEL) + LN_EPS);
  __bf16* orow = out + (size_t)row * D_MODEL;
#pragma unroll
  for (int i = 0; i < 4; ++i) {
    int c = tid + 256 * i;
    orow[c] = (__bf16)((v4[i] - mu) * inv * g[c] + be[c]);
  }
}

__global__ __launch_bounds__(256) void cast_bf16_kernel(const float* __restrict__ in, __bf16* __restrict__ out, int n) {
  for (int i = blockIdx.x * blockDim.x + threadIdx.x; i < n; i += gridDim.x * blockDim.x)
    out[i] = (__bf16)in[i];
}

// ---------------- BitNet projection GEMM: y = gamma * (A @ Wt^T) + bias ----------------
// Each wave: 32(M) x 64(N), K=1024 in 32-steps; 8 waves/block stack 256 rows.
// W tile (64x32, shared by all 8 waves) is double-buffered in LDS, filled with
// global_load_async_to_lds_b128 (ASYNCcnt).
// MODE 0: Q -> [b][h][t][d]   MODE 1: K -> [b][h][t][d]
// MODE 2: V -> [b][h][d][t] (transposed)   MODE 3: O -> f32 d_out = residual + y
template <int MODE>
__global__ __launch_bounds__(256) void gemm_bitnet(const __bf16* __restrict__ A, const __bf16* __restrict__ W,
                                                   const float* __restrict__ bias, const float* __restrict__ gamma,
                                                   const float* __restrict__ resid, void* __restrict__ Out,
                                                   int Trows) {
  __shared__ __bf16 wtile[2][64 * 32];   // 8 KB ping-pong
  const int wave = (int)(threadIdx.x >> 5);
  const int lane = (int)(threadIdx.x & 31);
  const int m0 = blockIdx.x * 256 + wave * 32;
  const int n0 = blockIdx.y * 64;
  const int wrow = (int)(threadIdx.x >> 2);        // 0..63
  const int wch = ((int)threadIdx.x & 3) * 8;      // 8 bf16 = 16B chunks

  v8f acc[2][4] = {{zero8(), zero8(), zero8(), zero8()},
                   {zero8(), zero8(), zero8(), zero8()}};
  const __bf16* Ab0 = A + (size_t)m0 * D_MODEL;
  const __bf16* Ab1 = Ab0 + (size_t)16 * D_MODEL;

  auto stage = [&](int buf, int k0) {
    const __bf16* gp = W + (size_t)(n0 + wrow) * D_MODEL + k0 + wch;
    __bf16* lp = &wtile[buf][wrow * 32 + wch];
#if USE_ASYNC_LDS
    __builtin_amdgcn_global_load_async_to_lds_b128(
        (__attribute__((address_space(1))) v4i*)gp,
        (__attribute__((address_space(3))) v4i*)lp, 0, 0);
#else
    *(uint4*)lp = *(const uint4*)gp;
#endif
  };

  stage(0, 0);
  for (int k0 = 0; k0 < D_MODEL; k0 += 32) {
    const int cur = (k0 >> 5) & 1;
#if USE_ASYNC_LDS
    __builtin_amdgcn_s_wait_asynccnt(0);
#endif
    __syncthreads();                 // wtile[cur] filled; wtile[cur^1] fully consumed
    if (k0 + 32 < D_MODEL) {
      stage(cur ^ 1, k0 + 32);
      __builtin_prefetch(Ab0 + (size_t)(lane & 15) * D_MODEL + (k0 + 32) + ((lane >> 4) * 8), 0, 0);
      __builtin_prefetch(Ab1 + (size_t)(lane & 15) * D_MODEL + (k0 + 32) + ((lane >> 4) * 8), 0, 0);
    }
    v16bf a0 = load_fragA(Ab0, D_MODEL, k0);
    v16bf a1 = load_fragA(Ab1, D_MODEL, k0);
#pragma unroll
    for (int j = 0; j < 4; ++j) {
      v16bf b = load_fragB(&wtile[cur][j * 16 * 32], 32, 0);
      acc[0][j] = wmma_bf16(a0, b, acc[0][j]);
      acc[1][j] = wmma_bf16(a1, b, acc[1][j]);
    }
  }

  const float g = gamma[0];
  const int n = lane & 15, h = lane >> 4;
#pragma unroll
  for (int mt = 0; mt < 2; ++mt) {
#pragma unroll
    for (int j = 0; j < 4; ++j) {
      const int c = n0 + j * 16 + n;
      const float bv = bias[c];
#pragma unroll
      for (int v = 0; v < 8; ++v) {
        const int mrow = m0 + mt * 16 + 8 * h + v;
        float y = g * acc[mt][j][v] + bv;
        if (MODE == 3) {
          ((float*)Out)[(size_t)mrow * D_MODEL + c] = resid[(size_t)mrow * D_MODEL + c] + y;
        } else {
          const int b = mrow / Trows, t = mrow % Trows;
          const int hh = c >> 6, d = c & 63;
          __bf16* o = (__bf16*)Out;
          if (MODE == 2)
            o[((size_t)(b * NUM_HEADS + hh) * HEAD_DIM + d) * (size_t)Trows + t] = (__bf16)y;
          else
            o[((size_t)(b * NUM_HEADS + hh) * (size_t)Trows + t) * HEAD_DIM + d] = (__bf16)y;
        }
      }
    }
  }
}

// ---------------- fused flash attention: one wave = 16 queries x D=64 ----------------
// 64 keys per iteration: 8 score WMMAs, 8 P@V WMMAs, 2 ones-column WMMAs (row-sum
// of probs via the matrix unit), one 4-step ds_bpermute max-butterfly per 8 rows.
// K fragments preloaded before the score WMMAs; V fragments issued right after the
// softmax math so they overlap the LDS round-trip and barriers.
__global__ __launch_bounds__(128) void attn_kernel(const __bf16* __restrict__ Q, const __bf16* __restrict__ Km,
                                                   const __bf16* __restrict__ Vt, __bf16* __restrict__ Op) {
  __shared__ __bf16 lds[4][16 * 64];
  const int wave = (int)(threadIdx.x >> 5);
  const int lane = (int)(threadIdx.x & 31);
  const int keyl = lane & 15, hl = lane >> 4;
  const int tile = blockIdx.x * 4 + wave;
  const int bh = tile >> 6;   // T_Q/16 == 64 query tiles per (b,h)
  const int qt = tile & 63;
  const int b = bh >> 4, h = bh & 15;
  const __bf16* Qb = Q + ((size_t)bh * T_Q + (size_t)qt * 16) * HEAD_DIM;
  const __bf16* Kb = Km + (size_t)bh * T_CTX * HEAD_DIM;
  const __bf16* Vb = Vt + (size_t)bh * HEAD_DIM * T_CTX;

  const v16bf qa0 = load_fragA(Qb, HEAD_DIM, 0);
  const v16bf qa1 = load_fragA(Qb, HEAD_DIM, 32);

  // B fragment of all-ones column 0: lanes with n==0 hold 1.0, others 0.
  v16bf onesB;
  {
    const __bf16 ob = (__bf16)((keyl == 0) ? 1.f : 0.f);
#pragma unroll
    for (int i = 0; i < 16; ++i) onesB[i] = ob;
  }

  v8f o0 = zero8(), o1 = zero8(), o2 = zero8(), o3 = zero8(), o4 = zero8();
  float m[8];
#pragma unroll
  for (int v = 0; v < 8; ++v) m[v] = -1e30f;

  __bf16* lp = lds[wave];

  for (int kt = 0; kt < T_CTX; kt += 64) {
    if (kt + 64 < T_CTX) {
      __builtin_prefetch(Kb + (size_t)(kt + 64 + 2 * lane) * HEAD_DIM, 0, 0);
      __builtin_prefetch(Vb + (size_t)lane * T_CTX + kt + 64, 0, 0);
      __builtin_prefetch(Vb + (size_t)(lane + 32) * T_CTX + kt + 64, 0, 0);
    }
    // ---- preload all K fragments for 64 keys, then run the score WMMA chain ----
    v16bf kf[8];
#pragma unroll
    for (int j = 0; j < 4; ++j) {
      const __bf16* Kj = Kb + (size_t)(kt + j * 16) * HEAD_DIM;
      kf[2 * j + 0] = load_fragB(Kj, HEAD_DIM, 0);
      kf[2 * j + 1] = load_fragB(Kj, HEAD_DIM, 32);
    }
    v8f s[4] = {zero8(), zero8(), zero8(), zero8()};
#pragma unroll
    for (int j = 0; j < 4; ++j) {
      s[j] = wmma_bf16(qa0, kf[2 * j + 0], s[j]);
      s[j] = wmma_bf16(qa1, kf[2 * j + 1], s[j]);
    }

    // ---- online softmax: one butterfly per 8 rows covers all 64 keys ----
    float ef[8], p[4][8];
#pragma unroll
    for (int v = 0; v < 8; ++v) {
      const float a0 = s[0][v] * ATT_SCALE, a1 = s[1][v] * ATT_SCALE;
      const float a2 = s[2][v] * ATT_SCALE, a3 = s[3][v] * ATT_SCALE;
      float mx = fmaxf(fmaxf(a0, a1), fmaxf(a2, a3));
      mx = fmaxf(mx, __shfl_xor(mx, 1, 32));   // stays inside 16-lane halves,
      mx = fmaxf(mx, __shfl_xor(mx, 2, 32));   // matching the C-matrix row grouping
      mx = fmaxf(mx, __shfl_xor(mx, 4, 32));
      mx = fmaxf(mx, __shfl_xor(mx, 8, 32));
      const float mn = fmaxf(m[v], mx);
      ef[v] = __expf(m[v] - mn);
      m[v] = mn;
      p[0][v] = __expf(a0 - mn);
      p[1][v] = __expf(a1 - mn);
      p[2][v] = __expf(a2 - mn);
      p[3][v] = __expf(a3 - mn);
    }

    // ---- issue V fragment loads now; they complete under the LDS round-trip ----
    v16bf vf[8];
#pragma unroll
    for (int f = 0; f < 4; ++f) {
      const __bf16* Vf = Vb + (size_t)f * 16 * T_CTX + kt;
      vf[2 * f + 0] = load_fragB(Vf, T_CTX, 0);
      vf[2 * f + 1] = load_fragB(Vf, T_CTX, 32);
    }

#pragma unroll
    for (int v = 0; v < 8; ++v) {
      o0[v] *= ef[v]; o1[v] *= ef[v]; o2[v] *= ef[v]; o3[v] *= ef[v]; o4[v] *= ef[v];
    }

    // C-layout -> A-layout conversion through LDS: [16 rows][64 keys] bf16
    __syncthreads();
#pragma unroll
    for (int j = 0; j < 4; ++j)
#pragma unroll
      for (int v = 0; v < 8; ++v)
        lp[(v + 8 * hl) * 64 + j * 16 + keyl] = (__bf16)p[j][v];
    __syncthreads();
    const v16bf pa0 = load_fragA(lp, 64, 0);
    const v16bf pa1 = load_fragA(lp, 64, 32);

    o0 = wmma_bf16(pa0, vf[0], o0); o0 = wmma_bf16(pa1, vf[1], o0);
    o1 = wmma_bf16(pa0, vf[2], o1); o1 = wmma_bf16(pa1, vf[3], o1);
    o2 = wmma_bf16(pa0, vf[4], o2); o2 = wmma_bf16(pa1, vf[5], o2);
    o3 = wmma_bf16(pa0, vf[6], o3); o3 = wmma_bf16(pa1, vf[7], o3);
    o4 = wmma_bf16(pa0, onesB, o4); o4 = wmma_bf16(pa1, onesB, o4);
  }

  // l for rows of this half lives in column n==0 -> lane hl*16; broadcast it.
  const int srcl = hl << 4;
  float il[8];
#pragma unroll
  for (int v = 0; v < 8; ++v) il[v] = 1.f / __shfl(o4[v], srcl, 32);

  const size_t rowbase = (size_t)b * T_Q + (size_t)qt * 16;
  v8f oo[4] = {o0, o1, o2, o3};
#pragma unroll
  for (int j = 0; j < 4; ++j)
#pragma unroll
    for (int v = 0; v < 8; ++v)
      Op[(rowbase + 8 * hl + v) * D_MODEL + h * HEAD_DIM + j * 16 + keyl] = (__bf16)(oo[j][v] * il[v]);
}

extern "C" void kernel_launch(void* const* d_in, const int* in_sizes, int n_in,
                              void* d_out, int out_size, void* d_ws, size_t ws_size,
                              hipStream_t stream) {
  (void)in_sizes; (void)n_in; (void)out_size; (void)ws_size;
  const float* x   = (const float*)d_in[0];
  const float* ctx = (const float*)d_in[1];
  const float* Wq  = (const float*)d_in[2];
  const float* bq  = (const float*)d_in[3];
  const float* Wk  = (const float*)d_in[4];
  const float* bk  = (const float*)d_in[5];
  const float* Wv  = (const float*)d_in[6];
  const float* bv  = (const float*)d_in[7];
  const float* Wo  = (const float*)d_in[8];
  const float* bo  = (const float*)d_in[9];
  const float* lng = (const float*)d_in[10];
  const float* lnb = (const float*)d_in[11];
  float* out = (float*)d_out;

  char* w = (char*)d_ws;
  size_t off = 0;
  auto take = [&](size_t bytes) -> char* {
    char* p = w + off;
    off += (bytes + 255) & ~(size_t)255;
    return p;
  };
  float*  gamma = (float*)take(4 * sizeof(float));
  float*  part  = (float*)take(4 * 256 * sizeof(float));
  __bf16* Wqq   = (__bf16*)take((size_t)D_MODEL * D_MODEL * 2);
  __bf16* Wkq   = (__bf16*)take((size_t)D_MODEL * D_MODEL * 2);
  __bf16* Wvq   = (__bf16*)take((size_t)D_MODEL * D_MODEL * 2);
  __bf16* Woq   = (__bf16*)take((size_t)D_MODEL * D_MODEL * 2);
  __bf16* xn    = (__bf16*)take((size_t)BATCH * T_Q * D_MODEL * 2);
  __bf16* ctxb  = (__bf16*)take((size_t)BATCH * T_CTX * D_MODEL * 2);
  __bf16* qb    = (__bf16*)take((size_t)BATCH * T_Q * D_MODEL * 2);
  __bf16* kb    = (__bf16*)take((size_t)BATCH * T_CTX * D_MODEL * 2);
  __bf16* vtb   = (__bf16*)take((size_t)BATCH * T_CTX * D_MODEL * 2);
  __bf16* attn  = (__bf16*)take((size_t)BATCH * T_Q * D_MODEL * 2);

  absmean_s1<<<dim3(256, 4), 256, 0, stream>>>(Wq, Wk, Wv, Wo, part);
  absmean_s2<<<dim3(4), 256, 0, stream>>>(part, gamma);
  quantize_kernel<<<dim3(1024, 4), 256, 0, stream>>>(Wq, Wk, Wv, Wo, gamma, Wqq, Wkq, Wvq, Woq);
  ln_cast_kernel<<<dim3(BATCH * T_Q), 256, 0, stream>>>(x, lng, lnb, xn);
  cast_bf16_kernel<<<dim3(8192), 256, 0, stream>>>(ctx, ctxb, BATCH * T_CTX * D_MODEL);

  gemm_bitnet<0><<<dim3((BATCH * T_Q) / 256, D_MODEL / 64), 256, 0, stream>>>(xn, Wqq, bq, gamma + 0, nullptr, qb, T_Q);
  gemm_bitnet<1><<<dim3((BATCH * T_CTX) / 256, D_MODEL / 64), 256, 0, stream>>>(ctxb, Wkq, bk, gamma + 1, nullptr, kb, T_CTX);
  gemm_bitnet<2><<<dim3((BATCH * T_CTX) / 256, D_MODEL / 64), 256, 0, stream>>>(ctxb, Wvq, bv, gamma + 2, nullptr, vtb, T_CTX);

  attn_kernel<<<dim3((BATCH * NUM_HEADS * (T_Q / 16)) / 4), 128, 0, stream>>>(qb, kb, vtb, attn);

  gemm_bitnet<3><<<dim3((BATCH * T_Q) / 256, D_MODEL / 64), 256, 0, stream>>>(attn, Woq, bo, gamma + 3, x, out, T_Q);
}
